// BasicGRU_53386443489964
// MI455X (gfx1250) — matrix operand order repaired and checked by reading, not compiled, not run on hardware
//
#include <hip/hip_runtime.h>
#include <hip/hip_bf16.h>

// ---------------- problem constants ----------------
#define B_   64
#define T_   2048
#define D_   256
#define U_   512
#define K_   (D_ + U_)          // 768 = fused K (x-projection + recurrent)
#define KT_  (K_ / 32)          // 24 k-tiles of 32 (bf16 WMMA K)
#define NWG  16                 // persistent workgroups (co-resident)
#define US   (U_ / NWG)         // 32 U-columns per workgroup
#define LDA  (K_ + 8)           // padded LDS A row stride (shorts) -> bank spread

#define SMEM_BYTES (B_ * LDA * 2)                   // 99328 B

// workspace layout (bytes)
#define OFF_CTR 0
#define OFF_HR  1024
#define OFF_HBF (OFF_HR + B_ * U_ * 2)              // bf16 copy of h state
#define OFF_W1  (OFF_HBF + B_ * U_ * 2)
#define OFF_W2  (OFF_W1 + (size_t)K_ * 2 * U_ * 2)
#define OFF_XB  (OFF_W2 + (size_t)K_ * U_ * 2)

typedef __attribute__((ext_vector_type(16))) __bf16       v16bf;
typedef __attribute__((ext_vector_type(8)))  float        v8f;
typedef __attribute__((ext_vector_type(4)))  unsigned int u32x4;

union FragU { v16bf bf; u32x4 q[2]; };

__device__ __forceinline__ unsigned short f32_to_bf16(float f) {
  unsigned u = __float_as_uint(f);
  unsigned r = 0x7FFFu + ((u >> 16) & 1u);           // round-to-nearest-even
  return (unsigned short)((u + r) >> 16);
}
__device__ __forceinline__ float bf16_to_f32(unsigned short h) {
  return __uint_as_float(((unsigned)h) << 16);
}
// raw v_rcp_f32 instead of IEEE divide
__device__ __forceinline__ float sigmoidf_(float x) {
  return __builtin_amdgcn_rcpf(1.0f + __expf(-x));
}
// tanh(x) = 1 - 2/(exp(2x)+1); exact at +/-inf limits
__device__ __forceinline__ float tanhf_(float x) {
  return 1.0f - 2.0f * __builtin_amdgcn_rcpf(__expf(2.0f * x) + 1.0f);
}

// async bf16 copy: global -> LDS (16B per lane), ASYNCcnt-tracked, no VGPR data path
__device__ __forceinline__ void async_copy_b128(const unsigned short* gsrc, unsigned short* ldst) {
  unsigned dst = (unsigned)(size_t)(void*)ldst;       // low 32 bits = LDS offset
  unsigned long long src = (unsigned long long)(size_t)gsrc;
  asm volatile("global_load_async_to_lds_b128 %0, %1, off" :: "v"(dst), "v"(src) : "memory");
}
__device__ __forceinline__ void async_wait0() {
  asm volatile("s_wait_asynccnt 0x0" ::: "memory");
}

// monotone-counter grid barrier (all NWG workgroups co-resident)
__device__ __forceinline__ void grid_barrier(unsigned* ctr, unsigned target) {
  __threadfence();
  __syncthreads();
  if (threadIdx.x == 0) {
    __hip_atomic_fetch_add(ctr, 1u, __ATOMIC_ACQ_REL, __HIP_MEMORY_SCOPE_AGENT);
    while (__hip_atomic_load(ctr, __ATOMIC_ACQUIRE, __HIP_MEMORY_SCOPE_AGENT) < target) {
      __builtin_amdgcn_s_sleep(2);
    }
  }
  __syncthreads();
}

// 16x16 WMMA tile over K_=768, software-pipelined (double-buffered fragments)
__device__ __forceinline__ v8f gemm_tile(const unsigned short* ap, const unsigned short* wp) {
  v8f acc = {};
  FragU a0, b0, a1, b1;
  a0.q[0] = *(const u32x4*)(ap);      a0.q[1] = *(const u32x4*)(ap + 16);
  b0.q[0] = *(const u32x4*)(wp);      b0.q[1] = *(const u32x4*)(wp + 8);
#pragma unroll
  for (int kt = 0; kt < KT_; ++kt) {
    if (kt + 1 < KT_) {
      const unsigned short* ap1 = ap + (kt + 1) * 32;
      const unsigned short* wp1 = wp + (kt + 1) * 512;
      a1.q[0] = *(const u32x4*)(ap1); a1.q[1] = *(const u32x4*)(ap1 + 16);
      b1.q[0] = *(const u32x4*)(wp1); b1.q[1] = *(const u32x4*)(wp1 + 8);
      __builtin_prefetch(wp + (kt + 2) * 512, 0, 1);
    }
    acc = __builtin_amdgcn_wmma_f32_16x16x32_bf16(false, a0.bf, false, b0.bf,
                                                  (short)0, acc, false, false);
    a0 = a1; b0 = b1;
  }
  return acc;
}

// ---------------- prep kernels ----------------
__global__ void init_ctr_kernel(unsigned* ctr) { if (threadIdx.x == 0) *ctr = 0u; }

// x [B][T][D] f32  ->  xbf [T][B][D] bf16 (time-major for contiguous per-step loads)
__global__ void xpose_kernel(const float* __restrict__ x, unsigned short* __restrict__ xbf) {
  size_t i = (size_t)blockIdx.x * 256 + threadIdx.x;
  int d = (int)(i & (D_ - 1));
  size_t tb = i >> 8;                                 // D_ = 256
  int b = (int)(tb & (B_ - 1));
  size_t t = tb >> 6;                                 // B_ = 64
  xbf[i] = f32_to_bf16(x[((size_t)b * T_ + t) * D_ + d]);
}

// Pack W = [Wtop(256 rows); Wbot(512 rows)] (K_ x ncols, f32) into exact WMMA
// B-fragment lane order: out[((nt*KT_+kt)*32+lane)*16 + i] = W[kt*32+(lane>>4)*8+i+(i>=8?8:0)][nt*16+(lane&15)]
__global__ void packw_kernel(const float* __restrict__ Wtop, const float* __restrict__ Wbot,
                             unsigned short* __restrict__ out, int ncols) {
  int tile = blockIdx.x;                 // = nt*KT_ + kt
  int lane = threadIdx.x;                // 32 threads
  int kt = tile % KT_;
  int nt = tile / KT_;
  int col   = nt * 16 + (lane & 15);
  int rbase = kt * 32 + (lane >> 4) * 8;
  unsigned short* dst = out + ((size_t)tile * 32 + lane) * 16;
#pragma unroll
  for (int i = 0; i < 16; ++i) {
    int row = rbase + i + (i >= 8 ? 8 : 0);
    float v = (row < D_) ? Wtop[(size_t)row * ncols + col]
                         : Wbot[(size_t)(row - D_) * ncols + col];
    dst[i] = f32_to_bf16(v);
  }
}

// ---------------- persistent GRU scan kernel ----------------
__global__ void gru_persistent_kernel(const unsigned short* __restrict__ xbf,
                                      const unsigned short* __restrict__ W1p,  // [Wk;Wrk] packed, 1024 cols
                                      const unsigned short* __restrict__ W2p,  // [Wu;Wur] packed, 512 cols
                                      const float* __restrict__ brk,
                                      const float* __restrict__ bur,
                                      unsigned short* __restrict__ hr_g,       // B x U bf16 (h*r exchange)
                                      unsigned short* __restrict__ hbf_g,      // B x U bf16 (h state copy)
                                      unsigned* __restrict__ ctr,
                                      float* __restrict__ h_g) {               // B x U f32 (= d_out, live h)
  extern __shared__ char smem[];
  unsigned short* Abuf = (unsigned short*)smem;       // [64][LDA] bf16: [x_t | h or h*r]

  const int tid     = threadIdx.x;
  const int lane    = tid & 31;
  const int wav     = tid >> 5;           // 8 waves
  const int wg      = blockIdx.x;         // 16 workgroups
  const int u0      = wg * US;            // this WG's U-slice
  const int lanelow = lane & 15;
  const int lanehi  = lane >> 4;
  const int m0      = (wav & 3) * 16;     // wave's M-tile
  const int nn      = wav >> 2;           // wave's local N-tile (0..1); same for r/z/candidate
  const int ucol    = u0 + nn * 16 + lanelow;

  unsigned epoch = 0;

  // h0 = 0 in both f32 state and bf16 mirror (each WG owns its slice)
  for (int i = tid; i < B_ * US; i += 256) {
    int b = i / US, u = u0 + (i % US);
    h_g[b * U_ + u] = 0.0f;
    hbf_g[b * U_ + u] = 0;
  }
  grid_barrier(ctr, (++epoch) * NWG);

  const unsigned short* ap = Abuf + (m0 + lanelow) * LDA + lanehi * 8;

  for (int t = 0; t < T_; ++t) {
    // ---- stage 1: async-copy A = [x_t | h_bf16] into LDS ----
    {
      const unsigned short* xs = xbf + (size_t)t * (B_ * D_);
      for (int c = tid; c < (B_ * D_) / 8; c += 256) {          // 8 chunks/thread
        int b = c >> 5, d8 = (c & 31) * 8;
        async_copy_b128(xs + b * D_ + d8, Abuf + b * LDA + d8);
      }
      for (int c = tid; c < (B_ * U_) / 8; c += 256) {          // 16 chunks/thread
        int b = c >> 6, u8 = (c & 63) * 8;
        async_copy_b128(hbf_g + b * U_ + u8, Abuf + b * LDA + D_ + u8);
      }
      async_wait0();
    }
    __syncthreads();

    // ---- stage 2: gate GEMMs. r-tile published as h*r; z-tile kept in VGPRs ----
    float zreg[8];
    {
      v8f racc = gemm_tile(ap, W1p + (size_t)(wg * 2 + nn) * KT_ * 512 + (size_t)lane * 16);
      v8f zacc = gemm_tile(ap, W1p + (size_t)(32 + wg * 2 + nn) * KT_ * 512 + (size_t)lane * 16);
      const float rbias = brk[ucol];
      const float zbias = brk[U_ + ucol];
#pragma unroll
      for (int v = 0; v < 8; ++v) {
        int row = m0 + v + 8 * lanehi;
        float r  = sigmoidf_(racc[v] + rbias);
        float hv = bf16_to_f32(Abuf[row * LDA + D_ + ucol]);
        hr_g[row * U_ + ucol] = f32_to_bf16(r * hv);            // publish h*r
        zreg[v] = sigmoidf_(zacc[v] + zbias);                   // z stays in VGPRs
      }
    }
    grid_barrier(ctr, (++epoch) * NWG);   // all h*r columns visible device-wide

    // ---- stage 3: async-swap h -> h*r in LDS A (x part unchanged) ----
    for (int c = tid; c < (B_ * U_) / 8; c += 256) {
      int b = c >> 6, u8 = (c & 63) * 8;
      async_copy_b128(hr_g + b * U_ + u8, Abuf + b * LDA + D_ + u8);
    }
    async_wait0();
    __syncthreads();

    // ---- stage 4: candidate GEMM + state update (same wave owns matching z) ----
    {
      v8f cacc = gemm_tile(ap, W2p + (size_t)(wg * 2 + nn) * KT_ * 512 + (size_t)lane * 16);
      const float cbias = bur[ucol];
#pragma unroll
      for (int v = 0; v < 8; ++v) {
        int row = m0 + v + 8 * lanehi;
        float ht = tanhf_(cacc[v] + cbias);
        float z  = zreg[v];
        float ho = h_g[row * U_ + ucol];
        float hn = (1.0f - z) * ho + z * ht;
        h_g[row * U_ + ucol]   = hn;                  // f32 state (= d_out)
        hbf_g[row * U_ + ucol] = f32_to_bf16(hn);     // bf16 mirror for next step's GEMM
      }
    }
    grid_barrier(ctr, (++epoch) * NWG);   // h_new visible before next step's reads
  }
}

// ---------------- host launcher ----------------
extern "C" void kernel_launch(void* const* d_in, const int* in_sizes, int n_in,
                              void* d_out, int out_size, void* d_ws, size_t ws_size,
                              hipStream_t stream) {
  const float* x   = (const float*)d_in[0];
  const float* Wk  = (const float*)d_in[1];
  const float* Wrk = (const float*)d_in[2];
  const float* brk = (const float*)d_in[3];
  const float* Wu  = (const float*)d_in[4];
  const float* Wur = (const float*)d_in[5];
  const float* bur = (const float*)d_in[6];
  float* h_out = (float*)d_out;                    // [B][U] — doubles as live h state

  char* ws = (char*)d_ws;
  unsigned*       ctr = (unsigned*)(ws + OFF_CTR);
  unsigned short* hr  = (unsigned short*)(ws + OFF_HR);
  unsigned short* hbf = (unsigned short*)(ws + OFF_HBF);
  unsigned short* W1p = (unsigned short*)(ws + OFF_W1);
  unsigned short* W2p = (unsigned short*)(ws + OFF_W2);
  unsigned short* xbf = (unsigned short*)(ws + OFF_XB);

  init_ctr_kernel<<<1, 32, 0, stream>>>(ctr);
  xpose_kernel<<<(int)(((size_t)T_ * B_ * D_) / 256), 256, 0, stream>>>(x, xbf);
  packw_kernel<<<(2 * U_ / 16) * KT_, 32, 0, stream>>>(Wk, Wrk, W1p, 2 * U_);
  packw_kernel<<<(U_ / 16) * KT_,     32, 0, stream>>>(Wu, Wur, W2p, U_);

  (void)hipFuncSetAttribute((const void*)gru_persistent_kernel,
                            hipFuncAttributeMaxDynamicSharedMemorySize, SMEM_BYTES);
  gru_persistent_kernel<<<NWG, 256, SMEM_BYTES, stream>>>(xbf, W1p, W2p, brk, bur,
                                                          hr, hbf, ctr, h_out);
}
